// MoELayer_18451179504170
// MI455X (gfx1250) — matrix operand order repaired and checked by reading, not compiled, not run on hardware
//
#include <hip/hip_runtime.h>

// ---------------------------------------------------------------------------
// MoE expert-choice layer for MI455X (gfx1250), wave32 + WMMA bf16.
// B=4, S=2048, D=1024, E=16, DFF=2048, CAP=256, RH=128
// ---------------------------------------------------------------------------

#define Bb   4
#define Ss   2048
#define Dd   1024
#define Ee   16
#define DFF  2048
#define CAP  256
#define RH   128

typedef __attribute__((ext_vector_type(16))) __bf16 v16bf;
typedef __attribute__((ext_vector_type(8)))  float  v8f;

__device__ __forceinline__ __bf16 f2bf(float f) { return (__bf16)f; }

__device__ __forceinline__ v8f wmma_bf16(v16bf a, v16bf b, v8f c) {
  // (neg_a, A, neg_b, B, c_mod, C, reuse_a, reuse_b)
  return __builtin_amdgcn_wmma_f32_16x16x32_bf16(false, a, false, b, (short)0, c,
                                                 false, false);
}

union FragU { v16bf v; uint4 q[2]; };

// A fragment from a row-major 16xLDA bf16 tile. Contiguous k order (k = 16*(lane/16) + s): each lane's
// 16 values are two contiguous 8-element runs -> two ds_load_b128.
#define LDA 40  // padded leading dim (80B rows: 16B aligned, bank-friendly)
__device__ __forceinline__ v16bf load_a_frag(const __bf16* As, int lane) {
  int row = lane & 15;
  int kb  = (lane >> 4) << 4;
  const __bf16* p = As + row * LDA + kb;
  FragU u;
  u.q[0] = *(const uint4*)p;         // k = kb .. kb+7
  u.q[1] = *(const uint4*)(p + 8);   // k = kb+8 .. kb+15
  return u.v;
}

// B fragment from fragment-ordered LDS: base points at [32][16] bf16 for one
// 32x16 n-tile; lane's 16 values are 32 contiguous bytes -> two ds_load_b128.
__device__ __forceinline__ v16bf load_b_fragF(const __bf16* base, int lane) {
  const __bf16* p = base + lane * 16;
  FragU u;
  u.q[0] = *(const uint4*)p;
  u.q[1] = *(const uint4*)(p + 8);
  return u.v;
}

// ---------------------------------------------------------------------------
// 1. Router MLP: logits[b,s,e] = silu(h @ rw1) @ rw2.  One block per token.
// ---------------------------------------------------------------------------
__global__ __launch_bounds__(RH) void router_kernel(
    const float* __restrict__ h, const float* __restrict__ rw1,
    const float* __restrict__ rw2, float* __restrict__ logits) {
  int tok = blockIdx.x;  // 0 .. B*S-1
  __shared__ float sh[Dd];
  __shared__ float hid[RH];
  const float* hp = h + (size_t)tok * Dd;
  for (int i = threadIdx.x; i < Dd; i += RH) sh[i] = hp[i];
  __syncthreads();
  int u = threadIdx.x;  // hidden unit 0..127
  float acc = 0.f;
#pragma unroll 4
  for (int d = 0; d < Dd; ++d) acc += sh[d] * rw1[(size_t)d * RH + u];
  hid[u] = acc / (1.f + __expf(-acc));  // SiLU
  __syncthreads();
  if (u < Ee) {
    float l = 0.f;
#pragma unroll 4
    for (int r = 0; r < RH; ++r) l += hid[r] * rw2[r * Ee + u];
    logits[(size_t)tok * Ee + u] = l;
  }
}

// ---------------------------------------------------------------------------
// 2. Expert-choice top-k: per (b,e), pick top-CAP tokens by logit (== by prob)
//    via 2048-key bitonic sort in LDS; weights = exp(l - lmax) normalized.
// ---------------------------------------------------------------------------
__global__ __launch_bounds__(256) void topk_kernel(
    const float* __restrict__ logits, int* __restrict__ idx_out,
    float* __restrict__ w_out) {
  int be = blockIdx.x;
  int b  = be >> 4;
  int e  = be & 15;
  __shared__ unsigned long long keys[Ss];
  __shared__ float red[256];
  const float* lp = logits + (size_t)b * Ss * Ee + e;
  for (int s = threadIdx.x; s < Ss; s += 256) {
    unsigned int bits = __float_as_uint(lp[(size_t)s * Ee]);
    bits = (bits & 0x80000000u) ? ~bits : (bits | 0x80000000u);  // order-preserving
    keys[s] = ((unsigned long long)bits << 32) | (unsigned int)(~(unsigned int)s);
  }
  __syncthreads();
  for (int k = 2; k <= Ss; k <<= 1) {
    for (int j = k >> 1; j > 0; j >>= 1) {
      for (int t = threadIdx.x; t < (Ss / 2); t += 256) {
        int i   = 2 * t - (t & (j - 1));
        int ixj = i + j;
        bool desc = ((i & k) == 0);
        unsigned long long a = keys[i], c = keys[ixj];
        if (desc ? (a < c) : (a > c)) { keys[i] = c; keys[ixj] = a; }
      }
      __syncthreads();
    }
  }
  // keys[0..255] = top-CAP descending; global max is keys[0].
  int t = threadIdx.x;
  unsigned long long kk = keys[t];
  unsigned int enc = (unsigned int)(kk >> 32);
  unsigned int ob  = (enc & 0x80000000u) ? (enc ^ 0x80000000u) : ~enc;
  float v = __uint_as_float(ob);
  int  sidx = (int)(~(unsigned int)(kk & 0xFFFFFFFFu));
  unsigned int enc0 = (unsigned int)(keys[0] >> 32);
  unsigned int ob0  = (enc0 & 0x80000000u) ? (enc0 ^ 0x80000000u) : ~enc0;
  float vmax = __uint_as_float(ob0);
  float w = __expf(v - vmax);
  red[t] = w;
  __syncthreads();
  for (int st = 128; st > 0; st >>= 1) {
    if (t < st) red[t] += red[t + st];
    __syncthreads();
  }
  float inv = 1.f / (red[0] + 1e-9f);
  idx_out[be * CAP + t] = sidx;
  w_out[be * CAP + t]   = w * inv;
}

// ---------------------------------------------------------------------------
// 3. Gather selected tokens -> bf16 dense A operand. One block per (be,row).
// ---------------------------------------------------------------------------
__global__ __launch_bounds__(256) void gather_kernel(
    const float* __restrict__ h, const int* __restrict__ idx,
    __bf16* __restrict__ Xbf) {
  int row = blockIdx.x;        // 0 .. B*E*CAP-1
  int be  = row >> 8;
  int b   = be >> 4;
  int tok = idx[row];
  const float* src = h + ((size_t)b * Ss + tok) * Dd;
  __bf16* dst = Xbf + (size_t)row * Dd;
  for (int i = threadIdx.x; i < Dd; i += 256) dst[i] = f2bf(src[i]);
}

// ---------------------------------------------------------------------------
// WMMA GEMM tiling: BM=128, BN=64, BK=32; 256 threads = 8 waves,
// wave w owns rows [16w,16w+16) x all 64 cols (4 wmma n-tiles).
// Double-buffered LDS, register prefetch, one barrier per K-step.
// ---------------------------------------------------------------------------
#define BM 128
#define BN 64
#define BK 32
#define ASZ (BM * LDA)  // A tile elements per buffer

// 4. Up projection: Hbf = bf16( silu(X@w1[e]) * (X@w2[e]) ), fused SwiGLU.
//    grid.x = (DFF/BN)*(CAP/BM) = 64, grid.y = B*E = 64
__global__ __launch_bounds__(256) void expert_up_kernel(
    const __bf16* __restrict__ Xbf, const float* __restrict__ w1,
    const float* __restrict__ w2, __bf16* __restrict__ Hbf) {
  __shared__ __align__(16) __bf16 As[2 * ASZ];            // 20 KB
  __shared__ __align__(16) __bf16 BsF[2 * 8 * 512];       // [buf][mat*4+ntile][lane][elem] 16 KB

  int be   = blockIdx.y;
  int e    = be & 15;
  int nblk = blockIdx.x & 31;   // DFF/BN = 32
  int mblk = blockIdx.x >> 5;   // CAP/BM = 2
  int tid  = threadIdx.x;
  int wave = tid >> 5;
  int lane = tid & 31;

  const __bf16* Xrow = Xbf + ((size_t)be * CAP + mblk * BM) * Dd;
  const float*  W1   = w1 + (size_t)e * Dd * DFF + nblk * BN;
  const float*  W2   = w2 + (size_t)e * Dd * DFF + nblk * BN;

  // A staging: thread copies rows arow and arow+64, one b128 each
  int arow = tid >> 2, aseg = tid & 3;
  // B staging: one (4k x 4n) unit per thread; 2 mats * 8 kq * 16 nq = 256
  int mat = tid >> 7;
  int kq4 = ((tid >> 4) & 7) * 4;  // k offset 0..28
  int nq4 = (tid & 15) * 4;        // n offset 0..60
  const float* Wst = mat ? W2 : W1;
  int bbase = (mat * 4 + (nq4 >> 4)) * 512 + ((nq4 & 15) + (kq4 & 16)) * 16 + (kq4 & 15);

  uint4 pa0, pa1;            // A prefetch
  float4 pr0, pr1, pr2, pr3; // B prefetch

  auto gload = [&](int kk) {
    const __bf16* xp = Xrow + (size_t)arow * Dd + kk + aseg * 8;
    pa0 = *(const uint4*)xp;
    pa1 = *(const uint4*)(xp + (size_t)64 * Dd);
    const float* wp = Wst + (size_t)(kk + kq4) * DFF + nq4;
    pr0 = *(const float4*)(wp);
    pr1 = *(const float4*)(wp + DFF);
    pr2 = *(const float4*)(wp + 2 * DFF);
    pr3 = *(const float4*)(wp + 3 * DFF);
  };
  auto lstore = [&](int buf) {
    __bf16* ap = &As[buf * ASZ + arow * LDA + aseg * 8];
    *(uint4*)ap = pa0;
    *(uint4*)(ap + 64 * LDA) = pa1;
    __bf16* bp = &BsF[buf * 8 * 512 + bbase];
#pragma unroll
    for (int j = 0; j < 4; ++j) {
      union { __bf16 h[4]; unsigned long long u; } p;
      p.h[0] = f2bf(((const float*)&pr0)[j]);
      p.h[1] = f2bf(((const float*)&pr1)[j]);
      p.h[2] = f2bf(((const float*)&pr2)[j]);
      p.h[3] = f2bf(((const float*)&pr3)[j]);
      *(unsigned long long*)&bp[j * 16] = p.u;
    }
  };

  v8f gacc[4], vacc[4];
#pragma unroll
  for (int nt = 0; nt < 4; ++nt) { v8f z = {}; gacc[nt] = z; vacc[nt] = z; }

  gload(0);
  lstore(0);
  __syncthreads();

  int buf = 0;
  for (int k0 = 0; k0 < Dd; k0 += BK) {
    if (k0 + BK < Dd) gload(k0 + BK);  // overlap global prefetch with math
    const __bf16* Ab = &As[buf * ASZ + (wave * 16) * LDA];
    const __bf16* Bbase = &BsF[buf * 8 * 512];
    v16bf af = load_a_frag(Ab, lane);
    v16bf bf1[4], bf2[4];
#pragma unroll
    for (int nt = 0; nt < 4; ++nt) {
      bf1[nt] = load_b_fragF(Bbase + nt * 512, lane);
      bf2[nt] = load_b_fragF(Bbase + (4 + nt) * 512, lane);
    }
#pragma unroll
    for (int nt = 0; nt < 4; ++nt) {
      gacc[nt] = wmma_bf16(af, bf1[nt], gacc[nt]);
      vacc[nt] = wmma_bf16(af, bf2[nt], vacc[nt]);
    }
    if (k0 + BK < Dd) lstore(buf ^ 1);
    __syncthreads();
    buf ^= 1;
  }

  // epilogue: silu(gate) * value -> bf16 H
  int crow = (lane >> 4) << 3;  // M offset 0 or 8
  int ccol = lane & 15;
  size_t Hbase = ((size_t)be * CAP + mblk * BM + wave * 16) * DFF + nblk * BN;
#pragma unroll
  for (int nt = 0; nt < 4; ++nt)
#pragma unroll
    for (int r = 0; r < 8; ++r) {
      float g  = gacc[nt][r];
      float hv = (g / (1.f + __expf(-g))) * vacc[nt][r];
      Hbf[Hbase + (size_t)(r + crow) * DFF + nt * 16 + ccol] = f2bf(hv);
    }
}

// 5. Down projection + weighted scatter-add: out[b,tok,:] += wt * (H @ w3[e]).
//    grid.x = (D/BN)*(CAP/BM) = 32, grid.y = B*E = 64
__global__ __launch_bounds__(256) void expert_down_kernel(
    const __bf16* __restrict__ Hbf, const float* __restrict__ w3,
    const int* __restrict__ tidx, const float* __restrict__ twts,
    float* __restrict__ out) {
  __shared__ __align__(16) __bf16 As[2 * ASZ];        // 20 KB
  __shared__ __align__(16) __bf16 BsF[2 * 4 * 512];   // [buf][ntile][lane][elem] 8 KB

  int be   = blockIdx.y;
  int b    = be >> 4;
  int e    = be & 15;
  int nblk = blockIdx.x & 15;   // D/BN = 16
  int mblk = blockIdx.x >> 4;   // CAP/BM = 2
  int tid  = threadIdx.x;
  int wave = tid >> 5;
  int lane = tid & 31;

  const __bf16* Hrow = Hbf + ((size_t)be * CAP + mblk * BM) * DFF;
  const float*  W3   = w3 + (size_t)e * DFF * Dd + nblk * BN;

  int arow = tid >> 2, aseg = tid & 3;
  // B staging: one (2k x 4n) unit per thread; 16 kp * 16 nq = 256
  int kp2 = ((tid >> 4) & 15) * 2;  // k offset 0..30
  int nq4 = (tid & 15) * 4;         // n offset 0..60
  int bbase = (nq4 >> 4) * 512 + ((nq4 & 15) + (kp2 & 16)) * 16 + (kp2 & 15);

  uint4 pa0, pa1;
  float4 pr0, pr1;

  auto gload = [&](int kk) {
    const __bf16* xp = Hrow + (size_t)arow * DFF + kk + aseg * 8;
    pa0 = *(const uint4*)xp;
    pa1 = *(const uint4*)(xp + (size_t)64 * DFF);
    const float* wp = W3 + (size_t)(kk + kp2) * Dd + nq4;
    pr0 = *(const float4*)(wp);
    pr1 = *(const float4*)(wp + Dd);
  };
  auto lstore = [&](int buf) {
    __bf16* ap = &As[buf * ASZ + arow * LDA + aseg * 8];
    *(uint4*)ap = pa0;
    *(uint4*)(ap + 64 * LDA) = pa1;
    __bf16* bp = &BsF[buf * 4 * 512 + bbase];
#pragma unroll
    for (int j = 0; j < 4; ++j) {
      union { __bf16 h[2]; unsigned int u; } p;
      p.h[0] = f2bf(((const float*)&pr0)[j]);
      p.h[1] = f2bf(((const float*)&pr1)[j]);
      *(unsigned int*)&bp[j * 16] = p.u;
    }
  };

  v8f acc[4];
#pragma unroll
  for (int nt = 0; nt < 4; ++nt) { v8f z = {}; acc[nt] = z; }

  gload(0);
  lstore(0);
  __syncthreads();

  int buf = 0;
  for (int k0 = 0; k0 < DFF; k0 += BK) {
    if (k0 + BK < DFF) gload(k0 + BK);
    const __bf16* Ab = &As[buf * ASZ + (wave * 16) * LDA];
    const __bf16* Bbase = &BsF[buf * 4 * 512];
    v16bf af = load_a_frag(Ab, lane);
    v16bf bfr[4];
#pragma unroll
    for (int nt = 0; nt < 4; ++nt) bfr[nt] = load_b_fragF(Bbase + nt * 512, lane);
#pragma unroll
    for (int nt = 0; nt < 4; ++nt) acc[nt] = wmma_bf16(af, bfr[nt], acc[nt]);
    if (k0 + BK < DFF) lstore(buf ^ 1);
    __syncthreads();
    buf ^= 1;
  }

  int crow = (lane >> 4) << 3;
  int ccol = lane & 15;
#pragma unroll
  for (int r = 0; r < 8; ++r) {
    int grow = mblk * BM + wave * 16 + r + crow;  // row within CAP
    int tok  = tidx[be * CAP + grow];
    float wt = twts[be * CAP + grow];
    float* op = out + ((size_t)b * Ss + tok) * Dd + nblk * BN;
#pragma unroll
    for (int nt = 0; nt < 4; ++nt)
      atomicAdd(&op[nt * 16 + ccol], acc[nt][r] * wt);
  }
}

// ---------------------------------------------------------------------------
// Host launcher
// ---------------------------------------------------------------------------
extern "C" void kernel_launch(void* const* d_in, const int* in_sizes, int n_in,
                              void* d_out, int out_size, void* d_ws, size_t ws_size,
                              hipStream_t stream) {
  const float* h   = (const float*)d_in[0];  // (B,S,D)
  const float* rw1 = (const float*)d_in[1];  // (D,128)
  const float* rw2 = (const float*)d_in[2];  // (128,E)
  const float* w1  = (const float*)d_in[3];  // (E,D,DFF)
  const float* w2  = (const float*)d_in[4];  // (E,D,DFF)
  const float* w3  = (const float*)d_in[5];  // (E,DFF,D)
  float* out = (float*)d_out;                // (B,S,D)

  // workspace carve-up (~97 MB)
  float*  logits = (float*)d_ws;                          // B*S*E      = 512 KB
  int*    tidx   = (int*)(logits + (size_t)Bb * Ss * Ee); // B*E*CAP    = 64 KB
  float*  twts   = (float*)(tidx + (size_t)Bb * Ee * CAP);// B*E*CAP    = 64 KB
  __bf16* Xbf    = (__bf16*)(twts + (size_t)Bb * Ee * CAP);   // 32 MB
  __bf16* Hbf    = Xbf + (size_t)Bb * Ee * CAP * Dd;          // 64 MB

  router_kernel<<<Bb * Ss, RH, 0, stream>>>(h, rw1, rw2, logits);
  topk_kernel<<<Bb * Ee, 256, 0, stream>>>(logits, tidx, twts);
  gather_kernel<<<Bb * Ee * CAP, 256, 0, stream>>>(h, tidx, Xbf);
  hipMemsetAsync(out, 0, (size_t)out_size * sizeof(float), stream);

  dim3 gUp((DFF / BN) * (CAP / BM), Bb * Ee);   // (64, 64)
  expert_up_kernel<<<gUp, 256, 0, stream>>>(Xbf, w1, w2, Hbf);

  dim3 gDn((Dd / BN) * (CAP / BM), Bb * Ee);    // (32, 64)
  expert_down_kernel<<<gDn, 256, 0, stream>>>(Hbf, w3, tidx, twts, out);
}
